// NoPriorVSSLModel_3891240370605
// MI455X (gfx1250) — matrix operand order repaired and checked by reading, not compiled, not run on hardware
//
#include <hip/hip_runtime.h>
#include <hip/hip_bf16.h>
#include <math.h>

#define Bsz 256
#define Csz 512
#define HWp 196           // 14*14
#define MAXIT 32
#define TAUc 0.03f
#define INVTAU (1.0f / TAUc)
#define EPSc 0.65f
#define EPS2c 0.40f
#define TLOG 0.07f
#define PDEPS 1e-6f

typedef __attribute__((ext_vector_type(16))) _Float16 v16h;
typedef __attribute__((ext_vector_type(8)))  _Float16 v8h;
typedef __attribute__((ext_vector_type(4)))  _Float16 v4h;
typedef __attribute__((ext_vector_type(8)))  float    v8f;
typedef __attribute__((ext_vector_type(4)))  float    v4f;

// single v_rcp_f32 instead of IEEE divide: this only feeds sigmoid weights
__device__ __forceinline__ float sigm(float x) {
    return __builtin_amdgcn_rcpf(1.0f + __expf(-x));
}

// ---------------------------------------------------------------------------
// Kernel A: pos_logits[b,p] = dot(img[b,:,p], aud[b,:]); also sim1[b], sim2[b]
// ---------------------------------------------------------------------------
__global__ __launch_bounds__(256) void pos_logits_kernel(
    const float* __restrict__ img, const float* __restrict__ aud,
    float* __restrict__ pos_out, float* __restrict__ sim1, float* __restrict__ sim2)
{
    __shared__ float audl[Csz];
    __shared__ float r0[256], r1[256], r2[256], r3[256];
    const int b = blockIdx.x, tid = threadIdx.x;
    for (int c = tid; c < Csz; c += 256) audl[c] = aud[b * Csz + c];
    __syncthreads();

    float s0 = 0.f, s1 = 0.f, s2 = 0.f, s3 = 0.f;
    if (tid < HWp) {
        const float* ip = img + (size_t)b * Csz * HWp + tid;
        float acc = 0.f;
        #pragma unroll 4
        for (int c = 0; c < Csz; ++c) acc = fmaf(ip[(size_t)c * HWp], audl[c], acc);
        pos_out[b * HWp + tid] = acc;
        float w   = sigm((acc - EPSc)  * INVTAU);
        float ngv = 1.0f - sigm((acc - EPS2c) * INVTAU);
        s0 = w * acc; s1 = w; s2 = ngv * acc; s3 = ngv;
    }
    r0[tid] = s0; r1[tid] = s1; r2[tid] = s2; r3[tid] = s3;
    __syncthreads();
    for (int s = 128; s > 0; s >>= 1) {
        if (tid < s) { r0[tid] += r0[tid+s]; r1[tid] += r1[tid+s];
                       r2[tid] += r2[tid+s]; r3[tid] += r3[tid+s]; }
        __syncthreads();
    }
    if (tid == 0) { sim1[b] = r0[0] / r1[0]; sim2[b] = r2[0] / r3[0]; }
}

// ---------------------------------------------------------------------------
// Kernel B: fused WMMA GEMM + sigmoid-weighted pixel reduction.
// all_logits[n,k,p] = sum_c img[n,c,p]*aud[k,c]
// Block = (n, mtile): per 32-ch k-step, stage A tile (16px x 32ch, scalar but
// branchless) and B tile (256col x 32ch, float4 -> v4h vectorized) in LDS;
// 8 waves x 2 N-tiles; fragments read as ds_load_b128.
// ---------------------------------------------------------------------------
__global__ __launch_bounds__(256) void sim_wmma_kernel(
    const float* __restrict__ img, const float* __restrict__ aud,
    float* __restrict__ numer, float* __restrict__ denom)
{
    __shared__ alignas(16) _Float16 As[16 * 32];     // [m][k]   1 KB
    __shared__ alignas(16) _Float16 Bs[256 * 32];    // [col][k] 16 KB

    const int n     = blockIdx.x / 13;
    const int mtile = blockIdx.x % 13;
    const int tid   = threadIdx.x;
    const int lane  = tid & 31;
    const int wave  = tid >> 5;          // 8 waves -> N-tiles {2*wave, 2*wave+1}
    const int l16   = lane & 15;
    const int hi    = lane >> 4;         // half-wave select
    const int mbase = mtile * 16;

    const float* imgb = img + (size_t)n * Csz * HWp;

    v8f acc0 = {}, acc1 = {};

    for (int k0 = 0; k0 < Csz; k0 += 32) {
        __syncthreads();
        // --- stage A tile: 512 floats, 2/thread, branchless OOB via cndmask ---
        #pragma unroll
        for (int r = 0; r < 2; ++r) {
            const int idx = tid + r * 256;           // idx = kc*16 + mm
            const int kc = idx >> 4, mm = idx & 15;
            const int m  = mbase + mm;
            const int mc = (m < HWp) ? m : (HWp - 1);
            const float av = imgb[(size_t)(k0 + kc) * HWp + mc];
            As[mm * 32 + kc] = (_Float16)((m < HWp) ? av : 0.0f);
            if (r == 0 && k0 + 32 < Csz)             // prefetch next k-chunk
                __builtin_prefetch(&imgb[(size_t)(k0 + 32 + kc) * HWp + mc], 0, 1);
        }
        // --- stage B tile: 2048 float4s, 8/thread, b128 loads + b64 stores ---
        #pragma unroll
        for (int r = 0; r < 8; ++r) {
            const int idx4 = tid + r * 256;          // idx4 = col*8 + kc4
            const int col = idx4 >> 3, kc4 = idx4 & 7;
            const v4f x = *(const v4f*)&aud[(size_t)col * Csz + k0 + kc4 * 4];
            v4h hx;
            hx[0] = (_Float16)x[0]; hx[1] = (_Float16)x[1];
            hx[2] = (_Float16)x[2]; hx[3] = (_Float16)x[3];
            *(v4h*)&Bs[col * 32 + kc4 * 4] = hx;
        }
        __syncthreads();

        // A fragment (16x32 MxK): lane holds K in [hi*16, hi*16+16)
        const v8h* arow = (const v8h*)(As + l16 * 32);
        const v16h afrag = __builtin_shufflevector(arow[hi * 2], arow[hi * 2 + 1],
                            0,1,2,3,4,5,6,7,8,9,10,11,12,13,14,15);

        // B fragments (32x16 KxN): lane holds K in [hi*16, hi*16+16)
        {
            const v8h* brow = (const v8h*)(Bs + (((wave * 2 + 0) * 16 + l16) * 32));
            const v16h bfrag = __builtin_shufflevector(brow[hi * 2], brow[hi * 2 + 1],
                                0,1,2,3,4,5,6,7,8,9,10,11,12,13,14,15);
            acc0 = __builtin_amdgcn_wmma_f32_16x16x32_f16(false, afrag, false, bfrag,
                                                          (short)0, acc0, false, false);
        }
        {
            const v8h* brow = (const v8h*)(Bs + (((wave * 2 + 1) * 16 + l16) * 32));
            const v16h bfrag = __builtin_shufflevector(brow[hi * 2], brow[hi * 2 + 1],
                                0,1,2,3,4,5,6,7,8,9,10,11,12,13,14,15);
            acc1 = __builtin_amdgcn_wmma_f32_16x16x32_f16(false, afrag, false, bfrag,
                                                          (short)0, acc1, false, false);
        }
    }

    // Epilogue: C/D layout -> element v = (m = mbase+v+8*hi, k = colbase+l16)
    // branchless: arithmetic mask instead of exec-divergent guard
    #pragma unroll
    for (int u = 0; u < 2; ++u) {
        const v8f acc = u ? acc1 : acc0;
        const int colbase = (wave * 2 + u) * 16;
        float num = 0.f, den = 0.f;
        #pragma unroll
        for (int v = 0; v < 8; ++v) {
            const int p = mbase + v + 8 * hi;
            const float valid = (p < HWp) ? 1.0f : 0.0f;
            const float a = acc[v];
            const float w = sigm((a - EPSc) * INVTAU) * valid;
            num = fmaf(w, a, num);
            den += w;
        }
        num += __shfl_xor(num, 16);
        den += __shfl_xor(den, 16);
        if (lane < 16) {
            atomicAdd(&numer[n * Bsz + colbase + l16], num);
            atomicAdd(&denom[n * Bsz + colbase + l16], den);
        }
    }
}

// ---------------------------------------------------------------------------
// Kernel C: sequential 32-iteration selection loop, one block per batch.
// logit[p,c] cached in LDS as f16, row-major per pixel (196 x 520-half rows,
// ~204KB -> CDNA5 320KB LDS). Hot distance sweep reads rows as b128 vectors.
// ---------------------------------------------------------------------------
#define LSTR 520
#define ITER_SMEM 212992
__global__ __launch_bounds__(256) void iter_kernel(
    const float* __restrict__ img, const float* __restrict__ pos_in,
    float* __restrict__ obj_out)
{
    extern __shared__ char smem[];
    _Float16* L   = (_Float16*)smem;                 // [196][LSTR] f16
    float* sel    = (float*)(smem + 204800);         // [512]
    float* posv   = (float*)(smem + 206848);         // [256]
    float* negl   = (float*)(smem + 207872);         // [256]
    float* d0     = (float*)(smem + 208896);         // [256]
    float* red    = (float*)(smem + 209920);         // [256]
    int*   redi   = (int*)  (smem + 210944);         // [256]
    int*   alr    = (int*)  (smem + 211968);         // [256]
    __shared__ int s_idx, s_end, s_cnt;

    const int b = blockIdx.x, tid = threadIdx.x;
    const float* imgb = img + (size_t)b * Csz * HWp;

    float pv = (tid < HWp) ? pos_in[b * HWp + tid] : -1e30f;
    posv[tid] = pv;
    negl[tid] = (tid < HWp) ? (1.0f - sigm((pv - EPS2c) * INVTAU)) : 0.0f;
    alr[tid]  = 0;
    if (tid == 0) s_end = 0;
    __syncthreads();

    // fill logit tile: coalesced global reads, transposed LDS writes
    if (tid < HWp) {
        #pragma unroll 4
        for (int c = 0; c < Csz; ++c)
            L[tid * LSTR + c] = (_Float16)(imgb[(size_t)c * HWp + tid] * pv);
    }
    __syncthreads();

    // neg_val[c] = mean_p logit[p,c]*neg[p]  (stored in sel[] temporarily)
    for (int c = tid; c < Csz; c += 256) {
        float acc = 0.f;
        for (int p = 0; p < HWp; ++p) acc = fmaf((float)L[p * LSTR + c], negl[p], acc);
        sel[c] = acc * (1.0f / (float)HWp);
    }
    __syncthreads();

    // dist_to_zero[p] = ||logit[p,:] - neg_val + 1e-6||  (b128 LDS reads)
    if (tid < HWp) {
        const v8h* row = (const v8h*)(L + tid * LSTR);
        float acc = 0.f;
        #pragma unroll 4
        for (int cc = 0; cc < Csz / 8; ++cc) {
            const v8h x = row[cc];
            #pragma unroll
            for (int j = 0; j < 8; ++j) {
                float v = (float)x[j] - sel[cc * 8 + j] + PDEPS;
                acc = fmaf(v, v, acc);
            }
        }
        d0[tid] = sqrtf(acc);
    }
    __syncthreads();

    for (int t = 0; t < MAXIT; ++t) {
        // argmax(positive)  (strict > keeps lowest index on ties)
        red[tid] = posv[tid]; redi[tid] = tid;
        __syncthreads();
        for (int s = 128; s > 0; s >>= 1) {
            if (tid < s && red[tid + s] > red[tid]) { red[tid] = red[tid + s]; redi[tid] = redi[tid + s]; }
            __syncthreads();
        }
        if (tid == 0) s_idx = redi[0];
        __syncthreads();
        const int idx  = s_idx;
        const int endp = s_end;      // endpoint state BEFORE this iter's update

        // sel = logit[idx,:]; obj[t,b,:] = endpoint ? 0 : sel
        for (int c = tid; c < Csz; c += 256) {
            float sv = (float)L[idx * LSTR + c];
            sel[c] = sv;
            obj_out[((size_t)t * Bsz + b) * Csz + c] = endp ? 0.0f : sv;
        }
        __syncthreads();

        // dist_to_logit + mask updates
        int pf = 0;
        if (tid < HWp) {
            const v8h* row = (const v8h*)(L + tid * LSTR);
            float acc = 0.f;
            #pragma unroll 4
            for (int cc = 0; cc < Csz / 8; ++cc) {
                const v8h x = row[cc];
                #pragma unroll
                for (int j = 0; j < 8; ++j) {
                    float v = (float)x[j] - sel[cc * 8 + j] + PDEPS;
                    acc = fmaf(v, v, acc);
                }
            }
            float dist = sqrtf(acc);
            pf = (dist < d0[tid]) && (!alr[tid]);
            if (pf) { alr[tid] = 1; posv[tid] = 0.0f; }
        }
        red[tid] = (float)pf;
        __syncthreads();
        for (int s = 128; s > 0; s >>= 1) { if (tid < s) red[tid] += red[tid + s]; __syncthreads(); }
        if (tid == 0) s_cnt = (int)red[0];
        __syncthreads();
        red[tid] = (tid < HWp) ? posv[tid] : -1e30f;
        __syncthreads();
        for (int s = 128; s > 0; s >>= 1) { if (tid < s) red[tid] = fmaxf(red[tid], red[tid + s]); __syncthreads(); }
        if (tid == 0) { if (s_cnt == 0 || red[0] < 0.1f) s_end = 1; }
        __syncthreads();
    }
}

// ---------------------------------------------------------------------------
// Kernel D: logits = [sim1, sim*mask, sim2]/T ; loss = mean CE(target=0)
// ---------------------------------------------------------------------------
__global__ __launch_bounds__(256) void loss_kernel(
    const float* __restrict__ numer, const float* __restrict__ denom,
    const float* __restrict__ sim1,  const float* __restrict__ sim2,
    float* __restrict__ loss_out)
{
    __shared__ float red[256];
    const int n = threadIdx.x;
    const float invT = 1.0f / TLOG;
    const float l0   = sim1[n] * invT;
    const float lend = sim2[n] * invT;
    float mx = fmaxf(l0, lend);
    for (int k = 0; k < Bsz; ++k) {
        float v = numer[n * Bsz + k] / denom[n * Bsz + k];
        if (k == n) v *= -99.0f;             // mask = 1 - 100*eye
        v *= invT;
        mx = fmaxf(mx, v);
    }
    float s = __expf(l0 - mx) + __expf(lend - mx);
    for (int k = 0; k < Bsz; ++k) {
        float v = numer[n * Bsz + k] / denom[n * Bsz + k];
        if (k == n) v *= -99.0f;
        v *= invT;
        s += __expf(v - mx);
    }
    red[n] = -(l0 - (mx + __logf(s)));
    __syncthreads();
    for (int st = 128; st > 0; st >>= 1) { if (n < st) red[n] += red[n + st]; __syncthreads(); }
    if (n == 0) loss_out[0] = red[0] * (1.0f / (float)Bsz);
}

// ---------------------------------------------------------------------------
extern "C" void kernel_launch(void* const* d_in, const int* in_sizes, int n_in,
                              void* d_out, int out_size, void* d_ws, size_t ws_size,
                              hipStream_t stream)
{
    const float* img = (const float*)d_in[0];   // [B,C,H,W]
    const float* aud = (const float*)d_in[1];   // [B,C]

    float* out      = (float*)d_out;
    float* loss_out = out;                       // [1]
    float* pos_out  = out + 1;                   // [B*HW]
    float* obj_out  = out + 1 + Bsz * HWp;       // [32*B*C]

    float* numer = (float*)d_ws;                 // [B*B]
    float* denom = numer + Bsz * Bsz;            // [B*B]
    float* sim1  = denom + Bsz * Bsz;            // [B]
    float* sim2  = sim1 + Bsz;                   // [B]

    hipMemsetAsync(numer, 0, sizeof(float) * 2 * Bsz * Bsz, stream);

    pos_logits_kernel<<<Bsz, 256, 0, stream>>>(img, aud, pos_out, sim1, sim2);

    // 256 batches * 13 M-tiles, 8 waves/block covering all 16 N-tiles
    sim_wmma_kernel<<<Bsz * 13, 256, 0, stream>>>(img, aud, numer, denom);

    hipFuncSetAttribute(reinterpret_cast<const void*>(iter_kernel),
                        hipFuncAttributeMaxDynamicSharedMemorySize, ITER_SMEM);
    iter_kernel<<<Bsz, 256, ITER_SMEM, stream>>>(img, pos_out, obj_out);

    loss_kernel<<<1, 256, 0, stream>>>(numer, denom, sim1, sim2, loss_out);
}
